// MixerModel_26379689132418
// MI455X (gfx1250) — matrix operand order repaired and checked by reading, not compiled, not run on hardware
//
#include <hip/hip_runtime.h>

#define B_DIM 2
#define L_DIM 4096
#define DMODEL 256
#define N_LAYER 4
#define D_INNER 512
#define D_STATE 16
#define D_CONV 4
#define DT_RANK 16
#define ROWS (B_DIM * L_DIM) /* 8192 */
#define LN_EPS 1e-5f

typedef __attribute__((ext_vector_type(16))) __bf16 v16bf;
typedef __attribute__((ext_vector_type(8)))  float v8f;

union FragCvt { uint4 u[2]; v16bf v; };

static __device__ __forceinline__ unsigned short f2bf(float f) {
  union { float f; unsigned int u; } c; c.f = f;
  unsigned int u = c.u;
  unsigned int r = u + 0x7FFFu + ((u >> 16) & 1u); // round-to-nearest-even
  return (unsigned short)(r >> 16);
}

// CDNA5 async copy: global -> LDS, 16B per lane, tracked by ASYNCcnt.
static __device__ __forceinline__ void async_b128(const void* g, const void* l) {
  asm volatile("global_load_async_to_lds_b128 %0, %1, off"
               :: "v"((unsigned int)(unsigned long long)l),
                  "v"((unsigned long long)g)
               : "memory");
}
#define WAIT_ASYNC_3() asm volatile("s_wait_asynccnt 0x3" ::: "memory")
#define WAIT_ASYNC_0() asm volatile("s_wait_asynccnt 0x0" ::: "memory")

// ---------------- elementwise helpers ----------------
__global__ void embed_kernel(const float* __restrict__ a, const float* __restrict__ p,
                             float* __restrict__ r, int n) {
  int i = blockIdx.x * blockDim.x + threadIdx.x;
  if (i < n) r[i] = a[i] + p[i];
}

__global__ void f2bf_kernel(const float* __restrict__ in, unsigned short* __restrict__ out, int n) {
  int i = blockIdx.x * blockDim.x + threadIdx.x;
  if (i < n) out[i] = f2bf(in[i]);
}

// ---------------- fused residual-add + LayerNorm (wave per row, wave32) ----------------
__global__ __launch_bounds__(256) void ln_kernel(
    float* __restrict__ resid, const float* __restrict__ addin,
    const float* __restrict__ w, const float* __restrict__ b,
    unsigned short* __restrict__ out_bf, float* __restrict__ out_f)
{
  const int lane = threadIdx.x & 31;
  const int wave = threadIdx.x >> 5;
  const int row  = blockIdx.x * 8 + wave;
  const size_t base = (size_t)row * DMODEL + lane * 8;
  float v[8];
  *(float4*)&v[0] = *(const float4*)&resid[base];
  *(float4*)&v[4] = *(const float4*)&resid[base + 4];
  if (addin) {
    float a[8];
    *(float4*)&a[0] = *(const float4*)&addin[base];
    *(float4*)&a[4] = *(const float4*)&addin[base + 4];
#pragma unroll
    for (int i = 0; i < 8; i++) v[i] += a[i];
    *(float4*)&resid[base]     = *(float4*)&v[0];
    *(float4*)&resid[base + 4] = *(float4*)&v[4];
  }
  float s = 0.f, sq = 0.f;
#pragma unroll
  for (int i = 0; i < 8; i++) { s += v[i]; sq += v[i] * v[i]; }
#pragma unroll
  for (int off = 16; off > 0; off >>= 1) {
    s  += __shfl_xor(s,  off, 32);
    sq += __shfl_xor(sq, off, 32);
  }
  const float mean = s * (1.f / DMODEL);
  const float var  = sq * (1.f / DMODEL) - mean * mean;
  const float rstd = rsqrtf(var + LN_EPS);
#pragma unroll
  for (int i = 0; i < 8; i++) {
    const int c = lane * 8 + i;
    const float o = (v[i] - mean) * rstd * w[c] + b[c];
    if (out_bf) out_bf[base + i] = f2bf(o);
    if (out_f)  out_f[base + i]  = o;
  }
}

// ---------------- bf16 WMMA GEMM: C[M,N] = A[M,K] * W[N,K]^T ----------------
// Double-buffered LDS, async global->LDS copies (ASYNCcnt), 4 waves x 2 waves tiling,
// each wave computes a 32x32 tile as 2x2 v_wmma_f32_16x16x32_bf16 fragments.
#define BLK_M 128
#define BLK_N 64
#define BLK_K 32
#define LDSTR 40 /* bf16 elems per LDS row: 80B, 16B-aligned, bank-spread */

__global__ __launch_bounds__(256) void gemm_bf16_wmma(
    const unsigned short* __restrict__ A,  // [M,K] bf16
    const unsigned short* __restrict__ W,  // [N,K] bf16
    float* __restrict__ C,                 // [M,N] f32
    int M, int N, int K)
{
  __shared__ __align__(16) unsigned short As[2][BLK_M * LDSTR];
  __shared__ __align__(16) unsigned short Ws[2][BLK_N * LDSTR];
  const int tid  = threadIdx.x;
  const int lane = tid & 31;
  const int wave = tid >> 5;
  const int bm = blockIdx.x * BLK_M;
  const int bn = blockIdx.y * BLK_N;
  const int wm = (wave >> 1) * 32;  // 4 waves over M
  const int wn = (wave & 1) * 32;   // 2 waves over N
  const int half = lane >> 4;
  const int l16  = lane & 15;

  // per-thread copy assignments (uniform 3 async issues per K-tile per wave)
  const int ar0 = tid >> 2,          ac = (tid & 3) * 8; // A rows 0..63
  const int ar1 = (tid + 256) >> 2;                      // A rows 64..127
  const int wr  = tid >> 2,          wc = (tid & 3) * 8;
  const int wrow = (bn + wr < N) ? (bn + wr) : (N - 1);  // clamp: cols >= N never stored

  v8f acc[2][2] = {};

  const int nk = K / BLK_K;
  // prologue: tile 0 -> buffer 0
  {
    const int k0 = 0;
    async_b128(&A[(size_t)(bm + ar0) * K + k0 + ac], &As[0][ar0 * LDSTR + ac]);
    async_b128(&A[(size_t)(bm + ar1) * K + k0 + ac], &As[0][ar1 * LDSTR + ac]);
    async_b128(&W[(size_t)wrow * K + k0 + wc],       &Ws[0][wr  * LDSTR + wc]);
  }

  for (int kt = 0; kt < nk; kt++) {
    const int cur = kt & 1;
    if (kt + 1 < nk) { // issue next tile into the other buffer, then drain current tile
      const int k0 = (kt + 1) * BLK_K;
      const int nxt = cur ^ 1;
      async_b128(&A[(size_t)(bm + ar0) * K + k0 + ac], &As[nxt][ar0 * LDSTR + ac]);
      async_b128(&A[(size_t)(bm + ar1) * K + k0 + ac], &As[nxt][ar1 * LDSTR + ac]);
      async_b128(&W[(size_t)wrow * K + k0 + wc],       &Ws[nxt][wr  * LDSTR + wc]);
      WAIT_ASYNC_3(); // in-order completion: only the 3 newest (next-tile) remain outstanding
    } else {
      WAIT_ASYNC_0();
    }
    __syncthreads();

    // A fragment: 16x32, lane=M row; elems 0..7 -> K=half*8..+7, elems 8..15 -> K=16+half*8..+7
    v16bf afrag[2], bfrag[2];
#pragma unroll
    for (int f = 0; f < 2; f++) {
      FragCvt c;
      const unsigned short* p = &As[cur][(wm + f * 16 + l16) * LDSTR + half * 8];
      c.u[0] = *reinterpret_cast<const uint4*>(p);
      c.u[1] = *reinterpret_cast<const uint4*>(p + 16);
      afrag[f] = c.v;
    }
    // B fragment: 32x16, lane=N col; elems 0..7 -> K=half*8..+7, elems 8..15 -> K=16+half*8..+7
#pragma unroll
    for (int f = 0; f < 2; f++) {
      FragCvt c;
      const unsigned short* p = &Ws[cur][(wn + f * 16 + l16) * LDSTR + half * 8];
      c.u[0] = *reinterpret_cast<const uint4*>(p);
      c.u[1] = *reinterpret_cast<const uint4*>(p + 16);
      bfrag[f] = c.v;
    }
#pragma unroll
    for (int fm = 0; fm < 2; fm++)
#pragma unroll
      for (int fn = 0; fn < 2; fn++)
        acc[fm][fn] = __builtin_amdgcn_wmma_f32_16x16x32_bf16(
            false, afrag[fm], false, bfrag[fn], (short)0, acc[fm][fn], false, false);
    __syncthreads(); // all waves done reading before this buffer is overwritten next iter
  }

  // C/D layout: VGPR r -> M = r + half*8; lane l16 -> N
#pragma unroll
  for (int fm = 0; fm < 2; fm++) {
#pragma unroll
    for (int fn = 0; fn < 2; fn++) {
      const int col = bn + wn + fn * 16 + l16;
      if (col < N) {
#pragma unroll
        for (int r = 0; r < 8; r++) {
          const int row = bm + wm + fm * 16 + half * 8 + r;
          C[(size_t)row * N + col] = acc[fm][fn][r];
        }
      }
    }
  }
}

// ---------------- depthwise causal conv (k=4) + SiLU ----------------
__global__ __launch_bounds__(256) void conv_silu_kernel(
    const float* __restrict__ xz, const float* __restrict__ cw,
    const float* __restrict__ cb, float* __restrict__ xcf,
    unsigned short* __restrict__ xcb)
{
  const int idx = blockIdx.x * blockDim.x + threadIdx.x; // over ROWS*D_INNER
  const int d  = idx & (D_INNER - 1);
  const int bt = idx >> 9;
  const int t  = bt & (L_DIM - 1);
  float s = cb[d];
#pragma unroll
  for (int j = 0; j < D_CONV; j++) {
    const int tt = t - (D_CONV - 1) + j;
    if (tt >= 0)
      s += xz[(size_t)(bt - (D_CONV - 1) + j) * (2 * D_INNER) + d] * cw[d * D_CONV + j];
  }
  const float o = s / (1.f + __expf(-s));
  xcf[idx] = o;
  xcb[idx] = f2bf(o);
}

// ---------------- selective scan (sequential over L), fused dt-proj/softplus/D-skip/SiLU gate ----------------
#define TCH 64
__global__ __launch_bounds__(256) void scan_kernel(
    const float* __restrict__ xdbl,   // [ROWS,48]  (dt 0:16 | B 16:32 | C 32:48)
    const float* __restrict__ xconv,  // [ROWS,512]
    const float* __restrict__ xz,     // [ROWS,1024] (z at +512)
    const float* __restrict__ dt_w,   // [512,16]
    const float* __restrict__ dt_b,   // [512]
    const float* __restrict__ A_log,  // [512,16]
    const float* __restrict__ D_skip, // [512]
    unsigned short* __restrict__ yg)  // [ROWS,512] bf16
{
  __shared__ float sb[TCH * 48];
  const int b = blockIdx.x >> 1;
  const int d = ((blockIdx.x & 1) << 8) + threadIdx.x;
  float Ar[D_STATE], dtw[D_STATE], h[D_STATE];
#pragma unroll
  for (int n = 0; n < D_STATE; n++) {
    Ar[n]  = -__expf(A_log[d * D_STATE + n]);
    dtw[n] = dt_w[d * D_STATE + n];
    h[n]   = 0.f;
  }
  const float dtb = dt_b[d], Dv = D_skip[d];
  for (int t0 = 0; t0 < L_DIM; t0 += TCH) {
    __syncthreads();
    for (int i = threadIdx.x; i < TCH * 48; i += 256)
      sb[i] = xdbl[(size_t)(b * L_DIM + t0) * 48 + i];
    __syncthreads();
    for (int tt = 0; tt < TCH; tt++) {
      const size_t row = (size_t)b * L_DIM + t0 + tt;
      const float* xd = &sb[tt * 48];
      float dt = dtb;
#pragma unroll
      for (int r = 0; r < DT_RANK; r++) dt += xd[r] * dtw[r];
      dt = (dt > 20.f) ? dt : log1pf(__expf(dt));
      const float xt = xconv[row * D_INNER + d];
      const float zt = xz[row * (2 * D_INNER) + D_INNER + d];
      const float dtx = dt * xt;
      float y = 0.f;
#pragma unroll
      for (int n = 0; n < D_STATE; n++) {
        const float dA = __expf(dt * Ar[n]);
        h[n] = dA * h[n] + dtx * xd[16 + n];
        y += h[n] * xd[32 + n];
      }
      y += Dv * xt;
      y *= zt / (1.f + __expf(-zt));
      yg[row * D_INNER + d] = f2bf(y);
    }
  }
}

// ---------------- host orchestration ----------------
extern "C" void kernel_launch(void* const* d_in, const int* in_sizes, int n_in,
                              void* d_out, int out_size, void* d_ws, size_t ws_size,
                              hipStream_t stream) {
  const float* input_ids  = (const float*)d_in[0];
  const float* pos        = (const float*)d_in[1];
  const float* norm_w     = (const float*)d_in[2];
  const float* norm_b     = (const float*)d_in[3];
  const float* in_proj_w  = (const float*)d_in[4];
  const float* conv_w     = (const float*)d_in[5];
  const float* conv_b     = (const float*)d_in[6];
  const float* x_proj_w   = (const float*)d_in[7];
  const float* dt_w       = (const float*)d_in[8];
  const float* dt_b       = (const float*)d_in[9];
  const float* A_log      = (const float*)d_in[10];
  const float* D_skip     = (const float*)d_in[11];
  const float* out_proj_w = (const float*)d_in[12];
  const float* normf_w    = (const float*)d_in[13];
  const float* normf_b    = (const float*)d_in[14];

  char* p = (char*)d_ws;
  auto carve = [&](size_t bytes) { char* q = p; p += (bytes + 255) & ~(size_t)255; return (void*)q; };
  float* resid            = (float*)carve((size_t)ROWS * DMODEL * 4);
  float* hidden           = (float*)carve((size_t)ROWS * DMODEL * 4);
  unsigned short* hn_bf   = (unsigned short*)carve((size_t)ROWS * DMODEL * 2);
  float* xz               = (float*)carve((size_t)ROWS * 2 * D_INNER * 4);
  float* xconv            = (float*)carve((size_t)ROWS * D_INNER * 4);
  unsigned short* xconv_bf= (unsigned short*)carve((size_t)ROWS * D_INNER * 2);
  float* xdbl             = (float*)carve((size_t)ROWS * 48 * 4);
  unsigned short* yg      = (unsigned short*)carve((size_t)ROWS * D_INNER * 2);
  unsigned short* win_bf  = (unsigned short*)carve((size_t)N_LAYER * 2 * D_INNER * DMODEL * 2);
  unsigned short* wx_bf   = (unsigned short*)carve((size_t)N_LAYER * 48 * D_INNER * 2);
  unsigned short* wout_bf = (unsigned short*)carve((size_t)N_LAYER * DMODEL * D_INNER * 2);

  int n;
  n = N_LAYER * 2 * D_INNER * DMODEL;
  f2bf_kernel<<<(n + 255) / 256, 256, 0, stream>>>(in_proj_w, win_bf, n);
  n = N_LAYER * 48 * D_INNER;
  f2bf_kernel<<<(n + 255) / 256, 256, 0, stream>>>(x_proj_w, wx_bf, n);
  n = N_LAYER * DMODEL * D_INNER;
  f2bf_kernel<<<(n + 255) / 256, 256, 0, stream>>>(out_proj_w, wout_bf, n);

  n = ROWS * DMODEL;
  embed_kernel<<<(n + 255) / 256, 256, 0, stream>>>(input_ids, pos, resid, n);

  for (int l = 0; l < N_LAYER; l++) {
    ln_kernel<<<ROWS / 8, 256, 0, stream>>>(
        resid, (l == 0) ? nullptr : hidden,
        norm_w + l * DMODEL, norm_b + l * DMODEL, hn_bf, nullptr);
    gemm_bf16_wmma<<<dim3(ROWS / BLK_M, (2 * D_INNER) / BLK_N), 256, 0, stream>>>(
        hn_bf, win_bf + (size_t)l * 2 * D_INNER * DMODEL, xz, ROWS, 2 * D_INNER, DMODEL);
    conv_silu_kernel<<<(ROWS * D_INNER) / 256, 256, 0, stream>>>(
        xz, conv_w + l * D_INNER * D_CONV, conv_b + l * D_INNER, xconv, xconv_bf);
    gemm_bf16_wmma<<<dim3(ROWS / BLK_M, (48 + BLK_N - 1) / BLK_N), 256, 0, stream>>>(
        xconv_bf, wx_bf + (size_t)l * 48 * D_INNER, xdbl, ROWS, 48, D_INNER);
    scan_kernel<<<B_DIM * 2, 256, 0, stream>>>(
        xdbl, xconv, xz,
        dt_w + l * D_INNER * DT_RANK, dt_b + l * D_INNER,
        A_log + l * D_INNER * D_STATE, D_skip + l * D_INNER, yg);
    gemm_bf16_wmma<<<dim3(ROWS / BLK_M, DMODEL / BLK_N), 256, 0, stream>>>(
        yg, wout_bf + (size_t)l * DMODEL * D_INNER, hidden, ROWS, DMODEL, D_INNER);
  }
  ln_kernel<<<ROWS / 8, 256, 0, stream>>>(resid, hidden, normf_w, normf_b,
                                          nullptr, (float*)d_out);
}
